// GMViT_31525059953262
// MI455X (gfx1250) — matrix operand order repaired and checked by reading, not compiled, not run on hardware
//
#include <hip/hip_runtime.h>
#include <hip/hip_bf16.h>
#include <math.h>
#include <stdint.h>

// ---------------- model dims ----------------
#define B_    1024
#define V_    20
#define C_    512
#define H_    8
#define DEPTH_ 6
#define G_    12
#define N1_   21
#define N2_   13

typedef __bf16 bf16_t;
typedef bf16_t v16bf __attribute__((ext_vector_type(16)));
typedef float  v8f   __attribute__((ext_vector_type(8)));

union FragB { unsigned u[8]; v16bf v; };

static __device__ inline unsigned short f2bfbits(float f) {
    bf16_t h = (bf16_t)f;
    return __builtin_bit_cast(unsigned short, h);
}
static __device__ inline float gelu_f(float x) {
    return 0.5f * x * (1.0f + erff(x * 0.70710678118654752440f));
}

// ---------------- f32 -> bf16 transpose conversion ----------------
// src: slice z is [K x N] f32, dst: slice z is [N x K] bf16
__global__ __launch_bounds__(256) void k_cvt_t(const float* __restrict__ src,
                                               bf16_t* __restrict__ dst,
                                               int K, int N) {
    __shared__ float tile[32][33];
    const size_t zoff = (size_t)blockIdx.z * K * N;
    const int k0 = blockIdx.y << 5, n0 = blockIdx.x << 5;
    const int tx = threadIdx.x & 31, ty = threadIdx.x >> 5;   // 32 x 8
    #pragma unroll
    for (int j = 0; j < 32; j += 8) {
        const int k = k0 + ty + j, n = n0 + tx;
        tile[ty + j][tx] = (k < K && n < N) ? src[zoff + (size_t)k * N + n] : 0.f;
    }
    __syncthreads();
    #pragma unroll
    for (int j = 0; j < 32; j += 8) {
        const int n = n0 + ty + j, k = k0 + tx;
        if (n < N && k < K) dst[zoff + (size_t)n * K + k] = (bf16_t)tile[tx][ty + j];
    }
}

// ---------------- generic WMMA bf16 GEMM ----------------
// Out[M,N] = act(A[M,K] @ W[K,N] + bias) + res   (bias/res optional)
// Wt is the PRE-TRANSPOSED weight: [N][K] bf16 (16B-aligned rows: K % 8 == 0).
// act: 0=none, 1=gelu(exact), 2=relu
// Requires: M % 64 == 0, K % 32 == 0.  N handled with guards.
// Block tile 64x128, 8 waves, each wave -> 16x64 (4 accumulators).
// B tile is DMA'd straight into LDS with global_load_async_to_lds_b128.
__global__ __launch_bounds__(256) void k_gemm(
    const float* __restrict__ A, const bf16_t* __restrict__ Wt,
    const float* __restrict__ bias, const float* __restrict__ res,
    float* __restrict__ Out, int M, int N, int K, int act)
{
    __shared__ unsigned short As[64][34];                                // [m][k]
    __shared__ __attribute__((aligned(16))) unsigned short Bs[128][48];  // [n][k], 96B rows
    const int tid  = threadIdx.x;
    const int lane = tid & 31, wave = tid >> 5;
    const int wm = wave & 3, wn = wave >> 2;          // 4 (m) x 2 (n) waves
    const int m0 = blockIdx.y << 6, n0 = blockIdx.x << 7;
    const int mloc = lane & 15, half = lane >> 4;

    const v8f vzero = {0.f,0.f,0.f,0.f,0.f,0.f,0.f,0.f};
    v8f acc[4] = {vzero, vzero, vzero, vzero};

    // A staging coords (8 f32 per thread)
    const int ar = (tid * 8) >> 5;   // 0..63
    const int ac = (tid * 8) & 31;   // multiple of 8
    // B async-DMA coords: two 16B chunks per thread (128 rows x 32 k)
    const int bn0 = tid >> 2;            // 0..63
    const int bn1 = bn0 + 64;            // 64..127
    const int bk  = (tid & 3) << 3;      // 0,8,16,24
    const unsigned ldsB0 = (unsigned)(uintptr_t)&Bs[bn0][bk];
    const unsigned ldsB1 = (unsigned)(uintptr_t)&Bs[bn1][bk];
    const bool ok0 = (n0 + bn0) < N;
    const bool ok1 = (n0 + bn1) < N;
    const unsigned gb0 = (unsigned)(((size_t)(n0 + bn0) * K + bk) * sizeof(bf16_t));
    const unsigned gb1 = (unsigned)(((size_t)(n0 + bn1) * K + bk) * sizeof(bf16_t));
    const unsigned long long wbase = (unsigned long long)(uintptr_t)Wt;

    for (int k0 = 0; k0 < K; k0 += 32) {
        // --- B tile: async DMA global -> LDS (no VGPR round trip) ---
        {
            const unsigned kbyte = (unsigned)(k0 * 2);
            if (ok0) {
                const unsigned go = gb0 + kbyte;
                asm volatile("global_load_async_to_lds_b128 %0, %1, %2"
                             :: "v"(ldsB0), "v"(go), "s"(wbase) : "memory");
            }
            if (ok1) {
                const unsigned go = gb1 + kbyte;
                asm volatile("global_load_async_to_lds_b128 %0, %1, %2"
                             :: "v"(ldsB1), "v"(go), "s"(wbase) : "memory");
            }
        }
        // --- A tile 64x32: load f32, convert to bf16, store to LDS ---
        {
            const float* src = A + (size_t)(m0 + ar) * K + (k0 + ac);
            float4 f0 = *(const float4*)src;
            float4 f1 = *(const float4*)(src + 4);
            unsigned short* d = &As[ar][ac];
            d[0] = f2bfbits(f0.x); d[1] = f2bfbits(f0.y);
            d[2] = f2bfbits(f0.z); d[3] = f2bfbits(f0.w);
            d[4] = f2bfbits(f1.x); d[5] = f2bfbits(f1.y);
            d[6] = f2bfbits(f1.z); d[7] = f2bfbits(f1.w);
            if (k0 + 32 < K) __builtin_prefetch(src + 32, 0, 0);
        }
        asm volatile("s_wait_asynccnt 0x0" ::: "memory");
        __syncthreads();

        // --- fragments + 4 WMMAs (one A-frag feeds 4 B-frags) ---
        FragB fa;
        #pragma unroll
        for (int r = 0; r < 8; ++r) {
            const int ka = (half << 4) + (r << 1);                         // A 16x32 layout (contiguous k = 16*half + s)
            fa.u[r] = *(const unsigned*)&As[(wm << 4) + mloc][ka];
        }
        #pragma unroll
        for (int s = 0; s < 4; ++s) {
            FragB fb;
            #pragma unroll
            for (int r = 0; r < 8; ++r) {
                const int kb = (half << 4) + (r << 1);                     // B 32x16 layout
                fb.u[r] = *(const unsigned*)&Bs[(wn << 6) + (s << 4) + mloc][kb];
            }
            acc[s] = __builtin_amdgcn_wmma_f32_16x16x32_bf16(false, fa.v, false, fb.v,
                                                             (short)0, acc[s], false, false);
        }
        __syncthreads();
    }

    // --- epilogue: C/D layout: VGPR r -> M = r + 8*half, N = lane&15 ---
    #pragma unroll
    for (int s = 0; s < 4; ++s) {
        const int n = n0 + (wn << 6) + (s << 4) + mloc;
        if (n >= N) continue;
        const float bv = bias ? bias[n] : 0.f;
        #pragma unroll
        for (int r = 0; r < 8; ++r) {
            const int m = m0 + (wm << 4) + r + (half << 3);
            float v0 = acc[s][r] + bv;
            if (act == 1) v0 = gelu_f(v0);
            else if (act == 2) v0 = fmaxf(v0, 0.f);
            if (res) v0 += res[(size_t)m * N + n];
            Out[(size_t)m * N + n] = v0;
        }
    }
}

// ---------------- fused (x + pos) + LayerNorm ----------------
// posmode: 0 = none, 1 = pos shared per token (pos[row % ntok]), 2 = per row
__global__ __launch_bounds__(128) void k_ln(
    const float* __restrict__ x, const float* __restrict__ pos, int posmode, int ntok,
    const float* __restrict__ g, const float* __restrict__ bta,
    float* __restrict__ hout, float* __restrict__ zout)
{
    const int row = blockIdx.x, tid = threadIdx.x;
    __shared__ float red[128];
    float vals[4];
    float s = 0.f;
    #pragma unroll
    for (int j = 0; j < 4; ++j) {
        const int c = tid + (j << 7);
        float v = x[(size_t)row * C_ + c];
        if (posmode == 1) v += pos[(size_t)(row % ntok) * C_ + c];
        else if (posmode == 2) v += pos[(size_t)row * C_ + c];
        vals[j] = v; s += v;
    }
    red[tid] = s; __syncthreads();
    for (int st = 64; st > 0; st >>= 1) { if (tid < st) red[tid] += red[tid + st]; __syncthreads(); }
    const float mean = red[0] * (1.f / C_);
    __syncthreads();
    float s2 = 0.f;
    #pragma unroll
    for (int j = 0; j < 4; ++j) { float d = vals[j] - mean; s2 += d * d; }
    red[tid] = s2; __syncthreads();
    for (int st = 64; st > 0; st >>= 1) { if (tid < st) red[tid] += red[tid + st]; __syncthreads(); }
    const float inv = rsqrtf(red[0] * (1.f / C_) + 1e-5f);
    #pragma unroll
    for (int j = 0; j < 4; ++j) {
        const int c = tid + (j << 7);
        if (hout) hout[(size_t)row * C_ + c] = vals[j];
        zout[(size_t)row * C_ + c] = (vals[j] - mean) * inv * g[c] + bta[c];
    }
}

// ---------------- per (batch, head) attention ----------------
__global__ __launch_bounds__(256) void k_attn(
    const float* __restrict__ qkv, const float* __restrict__ kmask,
    float* __restrict__ o, int ntok)
{
    const int bh = blockIdx.x;
    const int b  = bh / H_, h = bh % H_;
    const int tid = threadIdx.x;
    __shared__ float qs[N1_ * 64], ks[N1_ * 64], vs[N1_ * 64], ps[N1_ * N1_];

    const float* base = qkv + (size_t)b * ntok * (3 * C_);
    for (int i = tid; i < ntok * 64; i += 256) {
        const int n = i >> 6, d = i & 63;
        const size_t rb = (size_t)n * (3 * C_) + h * 64 + d;
        qs[i] = base[rb];
        ks[i] = base[rb + C_];
        vs[i] = base[rb + 2 * C_];
    }
    __syncthreads();
    for (int i = tid; i < ntok * ntok; i += 256) {
        const int n = i / ntok, m = i % ntok;
        float s = 0.f;
        #pragma unroll
        for (int d = 0; d < 64; ++d) s += qs[n * 64 + d] * ks[m * 64 + d];
        s *= 0.125f;   // hd^-0.5, hd = 64
        if (kmask && kmask[(size_t)b * ntok + m] == 0.f) s += -1e9f;
        ps[n * N1_ + m] = s;
    }
    __syncthreads();
    if (tid < ntok) {
        float mx = -1e30f;
        for (int m = 0; m < ntok; ++m) mx = fmaxf(mx, ps[tid * N1_ + m]);
        float sum = 0.f;
        for (int m = 0; m < ntok; ++m) { float e = expf(ps[tid * N1_ + m] - mx); ps[tid * N1_ + m] = e; sum += e; }
        const float inv = 1.f / sum;
        for (int m = 0; m < ntok; ++m) ps[tid * N1_ + m] *= inv;
    }
    __syncthreads();
    for (int i = tid; i < ntok * 64; i += 256) {
        const int n = i >> 6, d = i & 63;
        float s = 0.f;
        for (int m = 0; m < ntok; ++m) s += ps[n * N1_ + m] * vs[m * 64 + d];
        o[((size_t)b * ntok + n) * C_ + h * 64 + d] = s;
    }
}

// ---------------- pos-embed MLP: gelu(p@w1+b1)@w2+b2 ----------------
// map 0: dest row = row + 1 (encoder1 pos, 20 views)
// map 1: dest row = (row/G)*13 + 1 + row%G (encoder2 ypos)
__global__ __launch_bounds__(128) void k_posmlp(
    const float* __restrict__ p3, const float* __restrict__ w1, const float* __restrict__ b1,
    const float* __restrict__ w2, const float* __restrict__ b2,
    float* __restrict__ out, int map)
{
    const int row = blockIdx.x, tid = threadIdx.x;
    __shared__ float hbuf[128];
    const float x0 = p3[row * 3 + 0], x1 = p3[row * 3 + 1], x2 = p3[row * 3 + 2];
    float hsum = x0 * w1[0 * 128 + tid] + x1 * w1[1 * 128 + tid] + x2 * w1[2 * 128 + tid] + b1[tid];
    hbuf[tid] = gelu_f(hsum);
    __syncthreads();
    const size_t drow = (map == 0) ? (size_t)(row + 1)
                                   : (size_t)(row / G_) * N2_ + 1 + (row % G_);
    for (int c = tid; c < C_; c += 128) {
        float s = b2[c];
        for (int t = 0; t < 128; ++t) s += hbuf[t] * w2[t * C_ + c];
        out[drow * C_ + c] = s;
    }
}

// ---------------- small builders ----------------
__global__ __launch_bounds__(128) void k_bcast_row(float* __restrict__ dst,
                                                   const float* __restrict__ src,
                                                   size_t rowStride) {
    const size_t base = (size_t)blockIdx.x * rowStride;
    for (int c = threadIdx.x; c < C_; c += 128) dst[base + c] = src[c];
}

__global__ __launch_bounds__(128) void k_build_x0(const float* __restrict__ img,
                                                  const float* __restrict__ cls,
                                                  float* __restrict__ x) {
    const int row = blockIdx.x;
    const int b = row / N1_, n = row % N1_;
    for (int c = threadIdx.x; c < C_; c += 128)
        x[(size_t)row * C_ + c] = (n == 0) ? cls[c]
                                           : img[((size_t)b * V_ + (n - 1)) * C_ + c];
}

__global__ __launch_bounds__(128) void k_build_y(float* __restrict__ y, float* __restrict__ kmask,
                                                 const float* __restrict__ vg,
                                                 const float* __restrict__ mask,
                                                 const float* __restrict__ cls) {
    const int row = blockIdx.x;
    const int b = row / N2_, n = row % N2_;
    if (n == 0) {
        for (int c = threadIdx.x; c < C_; c += 128) y[(size_t)row * C_ + c] = cls[c];
        if (threadIdx.x == 0) kmask[row] = 1.f;
    } else {
        for (int c = threadIdx.x; c < C_; c += 128)
            y[(size_t)row * C_ + c] = vg[((size_t)b * G_ + n - 1) * C_ + c];
        if (threadIdx.x == 0) kmask[row] = mask[(size_t)b * G_ + n - 1];
    }
}

// sigmoid attention scores + Xv writeout
__global__ __launch_bounds__(128) void k_att_xv(const float* __restrict__ xn,
                                                const float* __restrict__ att_w,
                                                const float* __restrict__ att_b,
                                                float* __restrict__ att_out,
                                                float* __restrict__ xv_out) {
    const int r = blockIdx.x;             // 0 .. B*V-1
    const int b = r / V_, v = r % V_;
    const int tid = threadIdx.x;
    __shared__ float red[128];
    const float* xr = xn + ((size_t)b * N1_ + 1 + v) * C_;
    float s = 0.f;
    #pragma unroll
    for (int j = 0; j < 4; ++j) {
        const int c = tid + (j << 7);
        const float val = xr[c];
        xv_out[(size_t)r * C_ + c] = val;
        s += val * att_w[c];
    }
    red[tid] = s; __syncthreads();
    for (int st = 64; st > 0; st >>= 1) { if (tid < st) red[tid] += red[tid + st]; __syncthreads(); }
    if (tid == 0) att_out[r] = 1.f / (1.f + expf(-(red[0] + att_b[0])));
}

// group pooling: segment max of Xv, mean of pos3, occupancy mask
__global__ __launch_bounds__(256) void k_group_pool(const float* __restrict__ att,
                                                    const float* __restrict__ xn,
                                                    const float* __restrict__ vpos,
                                                    float* __restrict__ vg,
                                                    float* __restrict__ pg,
                                                    float* __restrict__ mask) {
    const int b = blockIdx.x, tid = threadIdx.x;
    __shared__ int bins[V_];
    __shared__ float cnt[G_];
    if (tid < V_) {
        int bi = (int)floorf(att[(size_t)b * V_ + tid] * (float)G_);
        bins[tid] = min(G_ - 1, max(0, bi));
    }
    __syncthreads();
    if (tid < G_) {
        int c = 0;
        for (int v = 0; v < V_; ++v) if (bins[v] == tid) ++c;
        cnt[tid] = (float)c;
        mask[(size_t)b * G_ + tid] = (c > 0) ? 1.f : 0.f;
    }
    __syncthreads();
    if (tid < G_ * 3) {
        const int g = tid / 3, d = tid % 3;
        float s = 0.f;
        for (int v = 0; v < V_; ++v) if (bins[v] == g) s += vpos[v * 3 + d];
        pg[((size_t)b * G_ + g) * 3 + d] = s / fmaxf(cnt[g], 1.f);
    }
    for (int c = tid; c < C_; c += 256) {
        for (int g = 0; g < G_; ++g) {
            float mx = -INFINITY; bool any = false;
            for (int v = 0; v < V_; ++v) {
                if (bins[v] == g) {
                    mx = fmaxf(mx, xn[((size_t)b * N1_ + 1 + v) * C_ + c]);
                    any = true;
                }
            }
            vg[((size_t)b * G_ + g) * C_ + c] = any ? mx : 0.f;
        }
    }
}

// masked head pool: y_max + concat feature
__global__ __launch_bounds__(256) void k_head_pool(const float* __restrict__ yn,
                                                   const float* __restrict__ mask,
                                                   float* __restrict__ ymax_out,
                                                   float* __restrict__ cf) {
    const int b = blockIdx.x, tid = threadIdx.x;
    for (int c = tid; c < C_; c += 256) {
        float mx = -INFINITY;
        for (int g = 0; g < G_; ++g)
            if (mask[(size_t)b * G_ + g] > 0.f)
                mx = fmaxf(mx, yn[((size_t)b * N2_ + 1 + g) * C_ + c]);
        ymax_out[(size_t)b * C_ + c] = mx;
        cf[(size_t)b * (2 * C_) + C_ + c] = mx;
        cf[(size_t)b * (2 * C_) + c] = yn[((size_t)b * N2_) * C_ + c];
    }
}

// batchnorm (inference) + optional raw output + relu output
__global__ __launch_bounds__(256) void k_bn(const float* __restrict__ x,
                                            const float* __restrict__ g, const float* __restrict__ bt,
                                            const float* __restrict__ m, const float* __restrict__ vv,
                                            int nf, float* __restrict__ raw, float* __restrict__ rel) {
    const size_t i = (size_t)blockIdx.x * 256 + threadIdx.x;
    if (i >= (size_t)B_ * nf) return;
    const int c = (int)(i % nf);
    const float val = (x[i] - m[c]) * rsqrtf(vv[c] + 1e-5f) * g[c] + bt[c];
    if (raw) raw[i] = val;
    rel[i] = fmaxf(val, 0.f);
}

// ---------------- input indices (flattened setup_inputs leaves) ----------------
enum {
    I_IMG = 0, I_VPOS,
    I_CLS1, I_CLSPOS1, I_CLS2, I_CLSPOS2,
    I_PE1W1, I_PE1B1, I_PE1W2, I_PE1B2,
    I_PE2W1, I_PE2B1, I_PE2W2, I_PE2B2,
    I_B1_LN1G, I_B1_LN1B, I_B1_LN2G, I_B1_LN2B,
    I_B1_QKV, I_B1_PROJ, I_B1_PROJB, I_B1_FC1, I_B1_FC1B, I_B1_FC2, I_B1_FC2B,
    I_B2_LN1G, I_B2_LN1B, I_B2_LN2G, I_B2_LN2B,
    I_B2_QKV, I_B2_PROJ, I_B2_PROJB, I_B2_FC1, I_B2_FC1B, I_B2_FC2, I_B2_FC2B,
    I_NORMG, I_NORMB, I_ATTW, I_ATTB,
    I_H1W, I_H1B, I_BN1G, I_BN1B, I_BN1M, I_BN1V,
    I_H2W, I_H2B, I_BN2G, I_BN2B, I_BN2M, I_BN2V,
    I_H3W, I_H3B
};

extern "C" void kernel_launch(void* const* d_in, const int* in_sizes, int n_in,
                              void* d_out, int out_size, void* d_ws, size_t ws_size,
                              hipStream_t stream) {
    (void)in_sizes; (void)n_in; (void)out_size; (void)ws_size;
    const float* in[64];
    for (int i = 0; i < 54 && i < n_in; ++i) in[i] = (const float*)d_in[i];

    float* out = (float*)d_out;
    const size_t SZ_IMG  = (size_t)B_ * V_ * C_;
    const size_t OFF_XV   = SZ_IMG;
    const size_t OFF_ATT  = OFF_XV + SZ_IMG;
    const size_t OFF_YMAX = OFF_ATT + (size_t)B_ * V_;
    const size_t OFF_FG   = OFF_YMAX + (size_t)B_ * C_;
    const size_t OFF_PRED = OFF_FG + (size_t)B_ * 512;

    // bump allocator over workspace
    char* ws = (char*)d_ws;
    size_t cur = 0;
    auto alloc = [&](size_t bytes) -> void* {
        void* p = ws + cur;
        cur += (bytes + 255) & ~(size_t)255;
        return p;
    };

    const int M1 = B_ * N1_;   // 21504
    const int M2 = B_ * N2_;   // 13312

    // bf16 transposed-weight buffers
    const size_t n_qkv = (size_t)DEPTH_ * C_ * 3 * C_;
    const size_t n_prj = (size_t)DEPTH_ * C_ * C_;
    const size_t n_fc1 = (size_t)DEPTH_ * C_ * 4 * C_;
    const size_t n_fc2 = (size_t)DEPTH_ * 4 * C_ * C_;
    bf16_t* bq1  = (bf16_t*)alloc(n_qkv * 2);
    bf16_t* bp1  = (bf16_t*)alloc(n_prj * 2);
    bf16_t* bf1a = (bf16_t*)alloc(n_fc1 * 2);
    bf16_t* bf1b = (bf16_t*)alloc(n_fc2 * 2);
    bf16_t* bq2  = (bf16_t*)alloc(n_qkv * 2);
    bf16_t* bp2  = (bf16_t*)alloc(n_prj * 2);
    bf16_t* bf2a = (bf16_t*)alloc(n_fc1 * 2);
    bf16_t* bf2b = (bf16_t*)alloc(n_fc2 * 2);
    bf16_t* bh1  = (bf16_t*)alloc((size_t)2 * C_ * 512 * 2);
    bf16_t* bh2  = (bf16_t*)alloc((size_t)512 * 256 * 2);
    bf16_t* bh3  = (bf16_t*)alloc((size_t)256 * 40 * 2);

    // activation buffers (encoder2 reuses encoder1 buffers: M2 < M1)
    float* pos1   = (float*)alloc((size_t)N1_ * C_ * 4);
    float* xA     = (float*)alloc((size_t)M1 * C_ * 4);
    float* xB     = (float*)alloc((size_t)M1 * C_ * 4);
    float* zbuf   = (float*)alloc((size_t)M1 * C_ * 4);
    float* qkvbuf = (float*)alloc((size_t)M1 * 3 * C_ * 4);
    float* obuf   = (float*)alloc((size_t)M1 * C_ * 4);
    float* midbuf = (float*)alloc((size_t)M1 * 4 * C_ * 4);
    float* vg     = (float*)alloc((size_t)B_ * G_ * C_ * 4);
    float* pg     = (float*)alloc((size_t)B_ * G_ * 3 * 4);
    float* maskb  = (float*)alloc((size_t)B_ * G_ * 4);
    float* ypos   = (float*)alloc((size_t)B_ * N2_ * C_ * 4);
    float* kmaskb = (float*)alloc((size_t)B_ * N2_ * 4);
    float* cf     = (float*)alloc((size_t)B_ * 2 * C_ * 4);
    float* g1buf  = (float*)alloc((size_t)B_ * 512 * 4);
    float* rel1   = (float*)alloc((size_t)B_ * 512 * 4);
    float* h2buf  = (float*)alloc((size_t)B_ * 256 * 4);
    float* rel2   = (float*)alloc((size_t)B_ * 256 * 4);

    // 1) img passthrough output
    hipMemcpyAsync(out, in[I_IMG], SZ_IMG * sizeof(float), hipMemcpyDeviceToDevice, stream);

    // 2) weight conversion f32 -> bf16, transposed ([K][N] -> [N][K])
    auto cvt = [&](const float* s, bf16_t* d, int K, int N, int depth) {
        dim3 gr((N + 31) / 32, (K + 31) / 32, depth);
        k_cvt_t<<<gr, 256, 0, stream>>>(s, d, K, N);
    };
    cvt(in[I_B1_QKV],  bq1,  C_, 3 * C_, DEPTH_);
    cvt(in[I_B1_PROJ], bp1,  C_, C_,     DEPTH_);
    cvt(in[I_B1_FC1],  bf1a, C_, 4 * C_, DEPTH_);
    cvt(in[I_B1_FC2],  bf1b, 4 * C_, C_, DEPTH_);
    cvt(in[I_B2_QKV],  bq2,  C_, 3 * C_, DEPTH_);
    cvt(in[I_B2_PROJ], bp2,  C_, C_,     DEPTH_);
    cvt(in[I_B2_FC1],  bf2a, C_, 4 * C_, DEPTH_);
    cvt(in[I_B2_FC2],  bf2b, 4 * C_, C_, DEPTH_);
    cvt(in[I_H1W], bh1, 2 * C_, 512, 1);
    cvt(in[I_H2W], bh2, 512, 256, 1);
    cvt(in[I_H3W], bh3, 256, 40, 1);

    // 3) encoder1 positional embedding (shared across batch)
    k_bcast_row<<<1, 128, 0, stream>>>(pos1, in[I_CLSPOS1], 0);
    k_posmlp<<<V_, 128, 0, stream>>>(in[I_VPOS], in[I_PE1W1], in[I_PE1B1],
                                     in[I_PE1W2], in[I_PE1B2], pos1, 0);

    // 4) build x0 = [cls_token_1 ; img views]
    k_build_x0<<<M1, 128, 0, stream>>>(in[I_IMG], in[I_CLS1], xA);

    // 5) encoders
    auto run_encoder = [&](float*& ping, float*& pong, const float* pos, int posmode,
                           int ntok, int Mrows, const float* kmask,
                           const bf16_t* wq, const bf16_t* wp, const bf16_t* wf1, const bf16_t* wf2,
                           const float* ln1g, const float* ln1b,
                           const float* ln2g, const float* ln2b,
                           const float* projb, const float* fc1b, const float* fc2b) {
        const int mt = Mrows / 64;
        for (int i = 0; i < DEPTH_; ++i) {
            k_ln<<<Mrows, 128, 0, stream>>>(ping, pos, posmode, ntok,
                                            ln1g + i * C_, ln1b + i * C_, pong, zbuf);
            k_gemm<<<dim3(3 * C_ / 128, mt), 256, 0, stream>>>(
                zbuf, wq + (size_t)i * C_ * 3 * C_, nullptr, nullptr,
                qkvbuf, Mrows, 3 * C_, C_, 0);
            k_attn<<<B_ * H_, 256, 0, stream>>>(qkvbuf, kmask, obuf, ntok);
            k_gemm<<<dim3(C_ / 128, mt), 256, 0, stream>>>(
                obuf, wp + (size_t)i * C_ * C_, projb + i * C_, pong,
                ping, Mrows, C_, C_, 0);
            k_ln<<<Mrows, 128, 0, stream>>>(ping, nullptr, 0, ntok,
                                            ln2g + i * C_, ln2b + i * C_, nullptr, zbuf);
            k_gemm<<<dim3(4 * C_ / 128, mt), 256, 0, stream>>>(
                zbuf, wf1 + (size_t)i * C_ * 4 * C_, fc1b + i * 4 * C_, nullptr,
                midbuf, Mrows, 4 * C_, C_, 1 /*gelu*/);
            k_gemm<<<dim3(C_ / 128, mt), 256, 0, stream>>>(
                midbuf, wf2 + (size_t)i * 4 * C_ * C_, fc2b + i * C_, ping,
                pong, Mrows, C_, 4 * C_, 0);
            float* t = ping; ping = pong; pong = t;
        }
    };

    float *ping = xA, *pong = xB;
    run_encoder(ping, pong, pos1, 1, N1_, M1, nullptr,
                bq1, bp1, bf1a, bf1b,
                in[I_B1_LN1G], in[I_B1_LN1B], in[I_B1_LN2G], in[I_B1_LN2B],
                in[I_B1_PROJB], in[I_B1_FC1B], in[I_B1_FC2B]);

    // 6) final LN of encoder1 -> xn (zbuf)
    k_ln<<<M1, 128, 0, stream>>>(ping, nullptr, 0, N1_, in[I_NORMG], in[I_NORMB], nullptr, zbuf);

    // 7) attention scores + Xv output
    k_att_xv<<<B_ * V_, 128, 0, stream>>>(zbuf, in[I_ATTW], in[I_ATTB],
                                          out + OFF_ATT, out + OFF_XV);

    // 8) group pooling
    k_group_pool<<<B_, 256, 0, stream>>>(out + OFF_ATT, zbuf, in[I_VPOS], vg, pg, maskb);

    // 9) encoder2 positional embedding + token build
    k_bcast_row<<<B_, 128, 0, stream>>>(ypos, in[I_CLSPOS2], (size_t)N2_ * C_);
    k_posmlp<<<B_ * G_, 128, 0, stream>>>(pg, in[I_PE2W1], in[I_PE2B1],
                                          in[I_PE2W2], in[I_PE2B2], ypos, 1);
    float *pingY = ping, *pongY = pong;   // encoder1 buffers are free now
    k_build_y<<<B_ * N2_, 128, 0, stream>>>(pingY, kmaskb, vg, maskb, in[I_CLS2]);

    // 10) encoder2
    run_encoder(pingY, pongY, ypos, 2, N2_, M2, kmaskb,
                bq2, bp2, bf2a, bf2b,
                in[I_B2_LN1G], in[I_B2_LN1B], in[I_B2_LN2G], in[I_B2_LN2B],
                in[I_B2_PROJB], in[I_B2_FC1B], in[I_B2_FC2B]);

    // 11) final LN of encoder2 -> yn (zbuf), masked pooling, concat feature
    k_ln<<<M2, 128, 0, stream>>>(pingY, nullptr, 0, N2_, in[I_NORMG], in[I_NORMB], nullptr, zbuf);
    k_head_pool<<<B_, 256, 0, stream>>>(zbuf, maskb, out + OFF_YMAX, cf);

    // 12) classification head
    k_gemm<<<dim3(512 / 128, B_ / 64), 256, 0, stream>>>(
        cf, bh1, in[I_H1B], nullptr, g1buf, B_, 512, 2 * C_, 0);
    k_bn<<<(B_ * 512 + 255) / 256, 256, 0, stream>>>(
        g1buf, in[I_BN1G], in[I_BN1B], in[I_BN1M], in[I_BN1V], 512, out + OFF_FG, rel1);
    k_gemm<<<dim3(256 / 128, B_ / 64), 256, 0, stream>>>(
        rel1, bh2, in[I_H2B], nullptr, h2buf, B_, 256, 512, 0);
    k_bn<<<(B_ * 256 + 255) / 256, 256, 0, stream>>>(
        h2buf, in[I_BN2G], in[I_BN2B], in[I_BN2M], in[I_BN2V], 256, nullptr, rel2);
    k_gemm<<<dim3(1, B_ / 64), 256, 0, stream>>>(
        rel2, bh3, in[I_H3B], nullptr, out + OFF_PRED, B_, 40, 256, 0);
}
